// AdaptiveCoverageAttention_14396730376232
// MI455X (gfx1250) — matrix operand order repaired and checked by reading, not compiled, not run on hardware
//
#include <hip/hip_runtime.h>
#include <cstdint>
#include <cstddef>

// ---------------------------------------------------------------------------
// AdaptiveCoverageAttention forward for MI455X (gfx1250, wave32, WMMA).
// bf16 WMMA (f32 accum) for all GEMMs; TDM async tile loads; flash softmax.
// ---------------------------------------------------------------------------

typedef __attribute__((ext_vector_type(16))) __bf16   v16bf;
typedef __attribute__((ext_vector_type(8)))  float    v8f;
typedef __attribute__((ext_vector_type(4)))  unsigned v4u;
typedef __attribute__((ext_vector_type(8)))  int      v8i;
typedef __attribute__((ext_vector_type(4)))  int      v4i;

__device__ __forceinline__ unsigned short f2bf(float f) {
  unsigned u = __float_as_uint(f);
  unsigned r = u + 0x7FFFu + ((u >> 16) & 1u);   // round-to-nearest-even
  return (unsigned short)(r >> 16);
}

// Issue one 2-D TENSOR_LOAD_TO_LDS: tileCols elements/row (bf16), tileRows
// rows, global row stride strideElems (bf16 elements). D# per ISA §8.3-8.5.
__device__ __forceinline__ void tdm_load_2d(unsigned ldsOff, const void* gptr,
                                            unsigned tileCols, unsigned tileRows,
                                            unsigned strideElems) {
  unsigned long long ga = (unsigned long long)(uintptr_t)gptr;
  v4u g0;
  g0[0] = 1u;                                        // count=1, user descriptor
  g0[1] = ldsOff;                                    // lds_addr (bytes)
  g0[2] = (unsigned)(ga & 0xFFFFFFFFu);              // global_addr[31:0]
  g0[3] = (unsigned)((ga >> 32) & 0x1FFFFFFu) | (2u << 30); // addr[56:32]|type=2
  v8i g1;
  g1[0] = (int)(1u << 16);                           // data_size=1 (2 bytes)
  g1[1] = (int)(0xFFFFu << 16);                      // tensor_dim0 lo16 (huge)
  g1[2] = (int)(0x7FFFu | (0xFFFFu << 16));          // dim0 hi16 | dim1 lo16
  g1[3] = (int)(0x7FFFu | (tileCols << 16));         // dim1 hi16 | tile_dim0
  g1[4] = (int)(tileRows & 0xFFFFu);                 // tile_dim1 | tile_dim2=0
  g1[5] = (int)strideElems;                          // tensor_dim0_stride[31:0]
  g1[6] = 0;                                         // stride hi | dim1_stride lo
  g1[7] = 0;
  v4i z4 = {0, 0, 0, 0};
  v8i z8 = {0, 0, 0, 0, 0, 0, 0, 0};
  __builtin_amdgcn_tensor_load_to_lds(g0, g1, z4, z4, z8, 0);
}

union Frag { unsigned u[8]; v16bf v; };

// ---------------------------------------------------------------------------
// Small prologue kernels
// ---------------------------------------------------------------------------
__global__ void cvt_bf16_kernel(const float* __restrict__ in,
                                unsigned short* __restrict__ out, int n) {
  int i = blockIdx.x * blockDim.x + threadIdx.x;
  if (i < n) out[i] = f2bf(in[i]);
}

// in [R][C] f32 -> out [C][R] bf16 (32x32 LDS tiles)
__global__ __launch_bounds__(256) void tcvt_kernel(const float* __restrict__ in,
                                                   unsigned short* __restrict__ out,
                                                   int R, int C) {
  __shared__ float tile[32][33];
  int c0 = blockIdx.x * 32, r0 = blockIdx.y * 32;
  int tx = threadIdx.x & 31, ty = threadIdx.x >> 5;
#pragma unroll
  for (int i = 0; i < 32; i += 8)
    tile[ty + i][tx] = in[(size_t)(r0 + ty + i) * C + c0 + tx];
  __syncthreads();
#pragma unroll
  for (int i = 0; i < 32; i += 8)
    out[(size_t)(c0 + ty + i) * R + r0 + tx] = f2bf(tile[tx][ty + i]);
}

__global__ void pool_kernel(const float* __restrict__ x, float* __restrict__ pooled) {
  int idx = blockIdx.x * blockDim.x + threadIdx.x;   // over B*D = 2048
  int b = idx >> 10, d = idx & 1023;
  float s = 0.f;
  for (int n = 0; n < 2048; ++n) s += x[((size_t)b * 2048 + n) * 1024 + d];
  pooled[idx] = s * (1.0f / 2048.0f);
}

__global__ __launch_bounds__(256) void gate_kernel(const float* __restrict__ pooled,
                                                   const float* __restrict__ w_fg1,
                                                   const float* __restrict__ b_fg1,
                                                   const float* __restrict__ w_fg2,
                                                   const float* __restrict__ b_fg2,
                                                   float* __restrict__ g) {
  __shared__ float red[256];
  int b = blockIdx.x, j = threadIdx.x;
  float acc = b_fg1[j];
  for (int d = 0; d < 1024; ++d) acc += pooled[b * 1024 + d] * w_fg1[d * 256 + j];
  float h1 = acc / (1.f + __expf(-acc));             // silu
  red[j] = h1 * w_fg2[j];
  __syncthreads();
  for (int s = 128; s > 0; s >>= 1) {
    if (j < s) red[j] += red[j + s];
    __syncthreads();
  }
  if (j == 0) {
    float z = red[0] + b_fg2[0];
    g[b] = 1.f / (1.f + __expf(-z));                 // sigmoid
  }
}

// bias[b][h][m] = g[b] * (silu(cov*w_ce1+b_ce1) @ w_ce2 + b_ce2)[h]
__global__ __launch_bounds__(256) void cov_bias_kernel(const float* __restrict__ coverage,
                                                       const float* __restrict__ w_ce1,
                                                       const float* __restrict__ b_ce1,
                                                       const float* __restrict__ w_ce2,
                                                       const float* __restrict__ b_ce2,
                                                       const float* __restrict__ g,
                                                       float* __restrict__ biasBHM) {
  __shared__ float h1s[256];
  int bn = blockIdx.x;                               // over B*N = 4096
  int b = bn >> 11, n = bn & 2047;
  int j = threadIdx.x;
  float c = coverage[bn];
  float a = c * w_ce1[j] + b_ce1[j];
  h1s[j] = a / (1.f + __expf(-a));
  __syncthreads();
  if (j < 16) {
    float acc = b_ce2[j];
    for (int k = 0; k < 256; ++k) acc += h1s[k] * w_ce2[k * 16 + j];
    biasBHM[((size_t)b * 16 + j) * 2048 + n] = g[b] * acc;
  }
}

// ---------------------------------------------------------------------------
// bf16 WMMA GEMM: C[M][Nn] = A[M][K] @ BT[Nn][K]^T (+bias). Tile 128x128,
// 8 waves, 16 rows/wave, K-step 32. A/B tiles staged via TDM.
// ---------------------------------------------------------------------------
__global__ __launch_bounds__(256) void gemm_bf16_kernel(
    const unsigned short* __restrict__ A, const unsigned short* __restrict__ BT,
    float* __restrict__ Cf, unsigned short* __restrict__ Cbf,
    const float* __restrict__ bias, int M, int Nn, int K) {
  __shared__ unsigned short ldsA[128 * 32];
  __shared__ unsigned short ldsB[128 * 32];
  const int tid = threadIdx.x;
  const int wave = tid >> 5, lane = tid & 31;
  const int laneN = lane & 15, half = lane >> 4;
  const int row0 = blockIdx.y * 128;
  const int col0 = blockIdx.x * 128;
  const int waveRow = wave * 16;

  v8f acc[8];
#pragma unroll
  for (int t = 0; t < 8; ++t) acc[t] = (v8f){};

  for (int k0 = 0; k0 < K; k0 += 32) {
    if (wave == 0) {
      tdm_load_2d((unsigned)(uintptr_t)ldsA, A + (size_t)row0 * K + k0, 32, 128, K);
      tdm_load_2d((unsigned)(uintptr_t)ldsB, BT + (size_t)col0 * K + k0, 32, 128, K);
      __builtin_amdgcn_s_wait_tensorcnt(0);
    }
    __syncthreads();

    Frag af;
    const int ar = waveRow + laneN;
#pragma unroll
    for (int v = 0; v < 8; ++v) {                    // A layout 16x32 bf16
      int kp = half * 16 + v * 2;
      af.u[v] = *(const unsigned*)&ldsA[ar * 32 + kp];
    }
#pragma unroll
    for (int t = 0; t < 8; ++t) {
      Frag bfr;
      const int bn = t * 16 + laneN;
#pragma unroll
      for (int v = 0; v < 8; ++v) {                  // B layout 32x16 bf16
        int kp = half * 16 + v * 2;
        bfr.u[v] = *(const unsigned*)&ldsB[bn * 32 + kp];
      }
      acc[t] = __builtin_amdgcn_wmma_f32_16x16x32_bf16(
          false, af.v, false, bfr.v, (short)0, acc[t], false, false);
    }
    __syncthreads();
  }

#pragma unroll
  for (int t = 0; t < 8; ++t) {
    const int col = col0 + t * 16 + laneN;
    float bv = bias ? bias[col] : 0.f;
#pragma unroll
    for (int v = 0; v < 8; ++v) {                    // C: row = v + 8*half
      int row = row0 + waveRow + v + 8 * half;
      float val = acc[t][v] + bv;
      if (Cf) Cf[(size_t)row * Nn + col] = val;
      else    Cbf[(size_t)row * Nn + col] = f2bf(val);
    }
  }
}

// ---------------------------------------------------------------------------
// Flash attention: block = (b, h, 128 query rows), 8 waves x 16 rows.
// S = scale*QK^T + bias[m]; online softmax; O = P@V. Key tiles of 64.
// ---------------------------------------------------------------------------
__global__ __launch_bounds__(256) void attn_kernel(
    const unsigned short* __restrict__ qkv,   // [B*N][3*D] bf16
    const float* __restrict__ biasBHM,        // [B][H][N]
    unsigned short* __restrict__ attnout) {   // [B*N][D] bf16
  constexpr int Nn = 2048, D3 = 3072, Dd = 1024, HD = 64, H = 16;
  __shared__ unsigned short ldsK[64 * 64];        // [m][hd]
  __shared__ unsigned short ldsV[64 * 64];        // [hd][m] (transposed)
  __shared__ unsigned short ldsP[8][16 * 64];     // per-wave P tile
  const int tid = threadIdx.x, wave = tid >> 5, lane = tid & 31;
  const int laneN = lane & 15, half = lane >> 4;
  const int b = blockIdx.z, h = blockIdx.y;
  const int row0 = blockIdx.x * 128;
  const int myRow = row0 + wave * 16 + laneN;
  const float scale = 0.125f;                      // 1/sqrt(64)

  Frag qf[2];
  {
    const unsigned short* qrow = qkv + (size_t)(b * Nn + myRow) * D3 + h * HD;
#pragma unroll
    for (int c = 0; c < 2; ++c)
#pragma unroll
      for (int v = 0; v < 8; ++v) {
        int kp = c * 32 + half * 16 + v * 2;
        qf[c].u[v] = *(const unsigned*)(qrow + kp);
      }
  }

  v8f O[4];
#pragma unroll
  for (int t = 0; t < 4; ++t) O[t] = (v8f){};
  float mrow[8], lrow[8];
#pragma unroll
  for (int v = 0; v < 8; ++v) { mrow[v] = -3.0e38f; lrow[v] = 0.f; }

  const float* biasRow = biasBHM + ((size_t)b * H + h) * Nn;

  for (int m0 = 0; m0 < Nn; m0 += 64) {
    if (wave == 0)                                   // TDM: 64 rows x 64 bf16
      tdm_load_2d((unsigned)(uintptr_t)ldsK,
                  qkv + (size_t)(b * Nn + m0) * D3 + Dd + h * HD, 64, 64, D3);
    for (int e = tid; e < 64 * 64; e += 256) {       // V staged transposed
      int m = e >> 6, hd = e & 63;
      ldsV[hd * 64 + m] = qkv[(size_t)(b * Nn + m0 + m) * D3 + 2 * Dd + h * HD + hd];
    }
    if (wave == 0) __builtin_amdgcn_s_wait_tensorcnt(0);
    __syncthreads();

    // ---- S = scale*Q@K^T + bias ----
    float s[4][8];
#pragma unroll
    for (int t = 0; t < 4; ++t) {
      v8f c = (v8f){};
#pragma unroll
      for (int kc = 0; kc < 2; ++kc) {
        Frag bfr;
#pragma unroll
        for (int v = 0; v < 8; ++v) {
          int kp = kc * 32 + half * 16 + v * 2;
          bfr.u[v] = *(const unsigned*)&ldsK[(t * 16 + laneN) * 64 + kp];
        }
        c = __builtin_amdgcn_wmma_f32_16x16x32_bf16(
            false, qf[kc].v, false, bfr.v, (short)0, c, false, false);
      }
      float bb = biasRow[m0 + t * 16 + laneN];
#pragma unroll
      for (int v = 0; v < 8; ++v) s[t][v] = c[v] * scale + bb;
    }

    // ---- online softmax (row = v + 8*half; reduce over 16-lane halves) ----
#pragma unroll
    for (int v = 0; v < 8; ++v) {
      float mx = fmaxf(fmaxf(s[0][v], s[1][v]), fmaxf(s[2][v], s[3][v]));
#pragma unroll
      for (int sh = 1; sh <= 8; sh <<= 1) mx = fmaxf(mx, __shfl_xor(mx, sh, 32));
      float newm = fmaxf(mrow[v], mx);
      float alpha = __expf(mrow[v] - newm);
      float sum = 0.f;
#pragma unroll
      for (int t = 0; t < 4; ++t) { s[t][v] = __expf(s[t][v] - newm); sum += s[t][v]; }
#pragma unroll
      for (int sh = 1; sh <= 8; sh <<= 1) sum += __shfl_xor(sum, sh, 32);
      lrow[v] = lrow[v] * alpha + sum;
      mrow[v] = newm;
#pragma unroll
      for (int t = 0; t < 4; ++t) O[t][v] *= alpha;
    }

    // ---- repack P (C layout) -> A layout via per-wave LDS tile ----
    unsigned short* P = &ldsP[wave][0];
#pragma unroll
    for (int t = 0; t < 4; ++t)
#pragma unroll
      for (int v = 0; v < 8; ++v)
        P[(v + 8 * half) * 64 + t * 16 + laneN] = f2bf(s[t][v]);
    asm volatile("s_wait_dscnt 0x0" ::: "memory");

    Frag pf[2];
#pragma unroll
    for (int c = 0; c < 2; ++c)
#pragma unroll
      for (int v = 0; v < 8; ++v) {
        int kp = c * 32 + half * 16 + v * 2;
        pf[c].u[v] = *(const unsigned*)&P[laneN * 64 + kp];
      }

    // ---- O += P @ V ----
#pragma unroll
    for (int t = 0; t < 4; ++t)
#pragma unroll
      for (int kc = 0; kc < 2; ++kc) {
        Frag bfr;
#pragma unroll
        for (int v = 0; v < 8; ++v) {
          int kp = kc * 32 + half * 16 + v * 2;
          bfr.u[v] = *(const unsigned*)&ldsV[(t * 16 + laneN) * 64 + kp];
        }
        O[t] = __builtin_amdgcn_wmma_f32_16x16x32_bf16(
            false, pf[kc].v, false, bfr.v, (short)0, O[t], false, false);
      }
    __syncthreads();
  }

#pragma unroll
  for (int v = 0; v < 8; ++v) {
    float inv = 1.0f / lrow[v];
    int row = row0 + wave * 16 + v + 8 * half;
#pragma unroll
    for (int t = 0; t < 4; ++t)
      attnout[(size_t)(b * Nn + row) * Dd + h * HD + t * 16 + laneN] =
          f2bf(O[t][v] * inv);
  }
}

// ---------------------------------------------------------------------------
extern "C" void kernel_launch(void* const* d_in, const int* in_sizes, int n_in,
                              void* d_out, int out_size, void* d_ws, size_t ws_size,
                              hipStream_t stream) {
  constexpr int B = 2, Nn = 2048, D = 1024, H = 16;
  constexpr int M = B * Nn;                          // 4096
  const float* x        = (const float*)d_in[0];
  const float* coverage = (const float*)d_in[1];
  const float* w_qkv    = (const float*)d_in[2];
  const float* w_out    = (const float*)d_in[3];
  const float* b_out    = (const float*)d_in[4];
  const float* w_ce1    = (const float*)d_in[5];
  const float* b_ce1    = (const float*)d_in[6];
  const float* w_ce2    = (const float*)d_in[7];
  const float* b_ce2    = (const float*)d_in[8];
  const float* w_fg1    = (const float*)d_in[9];
  const float* b_fg1    = (const float*)d_in[10];
  const float* w_fg2    = (const float*)d_in[11];
  const float* b_fg2    = (const float*)d_in[12];
  float* out = (float*)d_out;

  size_t off = 0;
  auto alloc = [&](size_t bytes) {
    void* p = (char*)d_ws + off;
    off = (off + bytes + 255) & ~(size_t)255;
    return p;
  };
  unsigned short* xbf    = (unsigned short*)alloc((size_t)M * D * 2);
  unsigned short* wqkvT  = (unsigned short*)alloc((size_t)3 * D * D * 2);
  unsigned short* woutT  = (unsigned short*)alloc((size_t)D * D * 2);
  unsigned short* qkvbf  = (unsigned short*)alloc((size_t)M * 3 * D * 2);
  unsigned short* attnbf = (unsigned short*)alloc((size_t)M * D * 2);
  float* pooled          = (float*)alloc((size_t)B * D * 4);
  float* gbuf            = (float*)alloc((size_t)B * 4);
  float* biasBHM         = (float*)alloc((size_t)B * H * Nn * 4);
  (void)ws_size; (void)n_in; (void)in_sizes; (void)out_size;

  // Precision conversion / weight transposition
  cvt_bf16_kernel<<<(M * D) / 256, 256, 0, stream>>>(x, xbf, M * D);
  tcvt_kernel<<<dim3(3 * D / 32, D / 32), 256, 0, stream>>>(w_qkv, wqkvT, D, 3 * D);
  tcvt_kernel<<<dim3(D / 32, D / 32), 256, 0, stream>>>(w_out, woutT, D, D);

  // Gate + coverage bias
  pool_kernel<<<(B * D) / 256, 256, 0, stream>>>(x, pooled);
  gate_kernel<<<B, 256, 0, stream>>>(pooled, w_fg1, b_fg1, w_fg2, b_fg2, gbuf);
  cov_bias_kernel<<<B * Nn, 256, 0, stream>>>(coverage, w_ce1, b_ce1, w_ce2, b_ce2,
                                              gbuf, biasBHM);

  // qkv = x @ w_qkv  (bf16 out)
  gemm_bf16_kernel<<<dim3(3 * D / 128, M / 128), 256, 0, stream>>>(
      xbf, wqkvT, nullptr, qkvbf, nullptr, M, 3 * D, D);

  // Flash attention with coverage bias
  attn_kernel<<<dim3(Nn / 128, H, B), 256, 0, stream>>>(qkvbf, biasBHM, attnbf);

  // out = attn_out @ w_out + b_out  (f32 out)
  gemm_bf16_kernel<<<dim3(D / 128, M / 128), 256, 0, stream>>>(
      attnbf, woutT, out, nullptr, b_out, M, D, D);
}
